// RelativeMultiheadAttention_39213051412526
// MI455X (gfx1250) — compile-verified, matrix operand order checked by reading, not run on hardware
//
#include <hip/hip_runtime.h>
#include <hip/hip_bf16.h>

// ---------------------------------------------------------------------------
// RelativeMultiheadAttention for gfx1250 (MI455X), wave32 + WMMA f16->f32.
// T=1024, B=8, E=1024, H=16, d=64, MAX_REL=16.
// ---------------------------------------------------------------------------

typedef _Float16 half_t;
typedef __attribute__((ext_vector_type(16))) _Float16 v16h;
typedef __attribute__((ext_vector_type(8)))  float    v8f;
typedef int v4i_gcc __attribute__((vector_size(16)));

union V16 { v16h v; half_t e[16]; };
union V8F { v8f  v; float  e[8];  };

#define WMMA_F32_F16(A, B, C) \
    __builtin_amdgcn_wmma_f32_16x16x32_f16(false, (A), false, (B), (short)0, (C), false, false)

#if defined(__has_builtin)
#if __has_builtin(__builtin_amdgcn_global_load_async_to_lds_b128)
#define HAVE_ASYNC_LDS 1
#endif
#endif

static constexpr int T_SEQ = 1024;
static constexpr int BSZ   = 8;
static constexpr int EMB   = 1024;
static constexpr int H     = 16;
static constexpr int HD    = 64;
static constexpr int NREL  = 33;           // 2*MAX_REL+1
static constexpr int M_ROWS = T_SEQ * BSZ; // 8192

// --- fragment loaders --------------------------------------------------------
// CDNA5 16-bit A operand (16x32): lane = m + 16*g ; VGPR v holds K pair
//   K = (v<4 ? 0 : 16) + g*8 + (v%4)*2 + e     (ISA layout, CDNA5 ISA)
// B operand (32x16) loaded symmetrically: lane = n + 16*g holds column n.

// Row-major source, K contiguous (stride = row stride in halfs).
__device__ inline v16h load_frag_rm(const half_t* __restrict__ base, int stride,
                                    int k0, int lane) {
    int m = lane & 15, g = (lane >> 4) & 1;
    const half_t* p = base + (size_t)m * stride + k0 + g * 8;
    V16 f;
#pragma unroll
    for (int v = 0; v < 8; ++v) {
        int kb = ((v & 4) ? 16 : 0) + (v & 3) * 2;
        f.e[2 * v]     = p[kb];
        f.e[2 * v + 1] = p[kb + 1];
    }
    return f.v;
}

// ---------------------------------------------------------------------------
// elementwise helpers
// ---------------------------------------------------------------------------
__global__ void f32_to_f16_kernel(const float* __restrict__ in,
                                  half_t* __restrict__ out, size_t n) {
    size_t i = (size_t)blockIdx.x * blockDim.x + threadIdx.x;
    if (i < n) out[i] = (half_t)in[i];
}

__global__ void zero_f32_kernel(float* __restrict__ p, size_t n) {
    size_t i = (size_t)blockIdx.x * blockDim.x + threadIdx.x;
    if (i < n) p[i] = 0.0f;
}

// qrel[bh][t][j] = sum_d q_scaled[bh][t][d] * relkey[j][d]
__global__ void qrel_kernel(const half_t* __restrict__ qbuf,
                            const float* __restrict__ relk,
                            float* __restrict__ qrel) {
    size_t idx = (size_t)blockIdx.x * blockDim.x + threadIdx.x;
    if (idx >= (size_t)BSZ * H * T_SEQ * NREL) return;
    int j = (int)(idx % NREL);
    size_t row = idx / NREL; // bh*1024 + t
    const half_t* q  = qbuf + row * HD;
    const float*  rk = relk + (size_t)j * HD;
    float s = 0.0f;
#pragma unroll 8
    for (int d = 0; d < HD; ++d) s += (float)q[d] * rk[d];
    qrel[idx] = s;
}

// ---------------------------------------------------------------------------
// Shared GEMM inner loop: acc[4] (16x64 tile) += A[16xK] · W[64xK]^T,
// K = EMB, ping-pong double buffering (no register rotation copies).
// ---------------------------------------------------------------------------
__device__ inline void gemm_accum_16x64(const half_t* __restrict__ Ab,
                                        const half_t* __restrict__ Wb,
                                        int lane, V8F acc[4]) {
    v16h a0 = load_frag_rm(Ab, EMB, 0, lane);
    v16h b0[4];
#pragma unroll
    for (int j = 0; j < 4; ++j)
        b0[j] = load_frag_rm(Wb + (size_t)j * 16 * EMB, EMB, 0, lane);

#pragma unroll 1
    for (int kk = 0; kk < EMB; kk += 64) {
        // stage buf1 (kk+32) while computing on buf0
        v16h a1 = load_frag_rm(Ab, EMB, kk + 32, lane);
        v16h b1[4];
#pragma unroll
        for (int j = 0; j < 4; ++j)
            b1[j] = load_frag_rm(Wb + (size_t)j * 16 * EMB, EMB, kk + 32, lane);
        if (kk + 128 <= EMB) __builtin_prefetch(Ab + kk + 128, 0, 1);
#pragma unroll
        for (int j = 0; j < 4; ++j) acc[j].v = WMMA_F32_F16(a0, b0[j], acc[j].v);
        // stage buf0 (kk+64) while computing on buf1
        if (kk + 64 < EMB) {
            a0 = load_frag_rm(Ab, EMB, kk + 64, lane);
#pragma unroll
            for (int j = 0; j < 4; ++j)
                b0[j] = load_frag_rm(Wb + (size_t)j * 16 * EMB, EMB, kk + 64, lane);
        }
#pragma unroll
        for (int j = 0; j < 4; ++j) acc[j].v = WMMA_F32_F16(a1, b1[j], acc[j].v);
    }
}

// ---------------------------------------------------------------------------
// QKV projection: C[m,n] = A[m,:]·W[n,:] + bias[n]; scatter to per-head f16
// buffers. A: [8192,1024] f16 (row m = t*8+b), W: [3072,1024] f16.
// ---------------------------------------------------------------------------
__global__ __launch_bounds__(256) void gemm_qkv_kernel(
    const half_t* __restrict__ A, const half_t* __restrict__ W,
    const float* __restrict__ bias,
    half_t* __restrict__ qbuf, half_t* __restrict__ kbuf,
    half_t* __restrict__ vbuf) {
    const int lane = threadIdx.x & 31;
    const int wid  = blockIdx.x * (blockDim.x >> 5) + (threadIdx.x >> 5);
    const int TN = (3 * EMB) / 64; // 48
    const int m0 = (wid / TN) * 16;
    const int n0 = (wid % TN) * 64;

    V8F acc[4];
#pragma unroll
    for (int j = 0; j < 4; ++j) acc[j].v = (v8f){0.f,0.f,0.f,0.f,0.f,0.f,0.f,0.f};

    gemm_accum_16x64(A + (size_t)m0 * EMB, W + (size_t)n0 * EMB, lane, acc);

    const int Nl = lane & 15;
    const int gM = (lane >> 4) << 3;
#pragma unroll
    for (int j = 0; j < 4; ++j) {
        int n = n0 + j * 16 + Nl;
#pragma unroll
        for (int r = 0; r < 8; ++r) {
            int m = m0 + r + gM;
            float val = acc[j].e[r] + bias[n];
            int t = m >> 3, b = m & 7;
            if (n < EMB) {                        // Q (apply d^-0.5 = 0.125)
                int h = n >> 6, d = n & 63;
                qbuf[(((size_t)(b * H + h)) * T_SEQ + t) * HD + d] =
                    (half_t)(val * 0.125f);
            } else if (n < 2 * EMB) {             // K
                int n2 = n - EMB;
                int h = n2 >> 6, d = n2 & 63;
                kbuf[(((size_t)(b * H + h)) * T_SEQ + t) * HD + d] = (half_t)val;
            } else {                              // V
                int n2 = n - 2 * EMB;
                int h = n2 >> 6, d = n2 & 63;
                vbuf[(((size_t)(b * H + h)) * T_SEQ + t) * HD + d] = (half_t)val;
            }
        }
    }
}

// ---------------------------------------------------------------------------
// Attention: 1 wave handles (head bh, 16-row t tile).
// scores[16][1024] in LDS -> +rel bias -> softmax -> avg atomics -> p.v WMMA
// with v tiles double-buffered through LDS via async global->LDS copies.
// ---------------------------------------------------------------------------
__device__ inline void stage_v_tile(const half_t* __restrict__ gsrc,
                                    half_t* lds_dst, int lane) {
    // 32x64 halfs = 4KB, contiguous in global; 8 x (32 lanes x 16B)
#if HAVE_ASYNC_LDS
#pragma unroll
    for (int c = 0; c < 8; ++c) {
        __builtin_amdgcn_global_load_async_to_lds_b128(
            (__attribute__((address_space(1))) v4i_gcc*)(gsrc + c * 256 + lane * 8),
            (__attribute__((address_space(3))) v4i_gcc*)(lds_dst + c * 256 + lane * 8),
            0, 0);
    }
#else
#pragma unroll
    for (int c = 0; c < 8; ++c) {
        *(float4*)(lds_dst + c * 256 + lane * 8) =
            *(const float4*)(gsrc + c * 256 + lane * 8);
    }
#endif
}

__device__ inline void wait_v_tile(int pending_tiles) {
#if HAVE_ASYNC_LDS
    if (pending_tiles)
        asm volatile("s_wait_asynccnt 0x8" ::: "memory");
    else
        asm volatile("s_wait_asynccnt 0x0" ::: "memory");
#else
    (void)pending_tiles;
    __syncthreads();
#endif
}

// score-tile epilogue: add relative-position bias, spill to LDS stripe
__device__ inline void store_score_tile(float* __restrict__ sc, const V8F& acc,
                                        int t0, int s0, int gM, int Nl,
                                        const float* __restrict__ qrel_bh) {
#pragma unroll
    for (int r = 0; r < 8; ++r) {
        int M = r + gM;
        int t = t0 + M, s = s0 + Nl;
        int dlt = s - t;
        dlt = dlt < -16 ? -16 : (dlt > 16 ? 16 : dlt);
        sc[M * T_SEQ + s] = acc.e[r] + qrel_bh[(size_t)t * NREL + (dlt + 16)];
    }
}

__global__ __launch_bounds__(32) void attn_kernel(
    const half_t* __restrict__ qbuf, const half_t* __restrict__ kbuf,
    const half_t* __restrict__ vbuf, const float* __restrict__ qrel,
    float* __restrict__ avg, half_t* __restrict__ attnh) {
    __shared__ float sc[16 * T_SEQ];                    // 64 KB
    __shared__ __align__(16) half_t vtile[2][32 * HD];  // 8 KB

    const int lane = threadIdx.x;
    const int tt = blockIdx.x & 63;
    const int bh = blockIdx.x >> 6;
    const int b = bh >> 4, h = bh & 15;
    const int t0 = tt * 16;

    const half_t* qb = qbuf + (size_t)bh * T_SEQ * HD;
    const half_t* kb = kbuf + (size_t)bh * T_SEQ * HD;
    const half_t* vb = vbuf + (size_t)bh * T_SEQ * HD;
    const float*  qrel_bh = qrel + (size_t)bh * T_SEQ * NREL;

    // q tile fragments reused across the whole s sweep
    const v16h a0 = load_frag_rm(qb + (size_t)t0 * HD, HD, 0, lane);
    const v16h a1 = load_frag_rm(qb + (size_t)t0 * HD, HD, 32, lane);

    const int Nl = lane & 15;
    const int gM = (lane >> 4) << 3;
    const v8f zacc = (v8f){0.f,0.f,0.f,0.f,0.f,0.f,0.f,0.f};

    // ---- scores = q k^T + q rk^T (k tiles ping-pong double-buffered) ----
    v16h k0a = load_frag_rm(kb, HD, 0, lane);
    v16h k0b = load_frag_rm(kb, HD, 32, lane);
#pragma unroll 1
    for (int st = 0; st < 64; st += 2) {
        v16h k1a = load_frag_rm(kb + (size_t)(st + 1) * 16 * HD, HD, 0, lane);
        v16h k1b = load_frag_rm(kb + (size_t)(st + 1) * 16 * HD, HD, 32, lane);
        V8F acc;
        acc.v = WMMA_F32_F16(a0, k0a, zacc);
        acc.v = WMMA_F32_F16(a1, k0b, acc.v);
        store_score_tile(sc, acc, t0, st * 16, gM, Nl, qrel_bh);
        if (st + 2 < 64) {
            k0a = load_frag_rm(kb + (size_t)(st + 2) * 16 * HD, HD, 0, lane);
            k0b = load_frag_rm(kb + (size_t)(st + 2) * 16 * HD, HD, 32, lane);
        }
        acc.v = WMMA_F32_F16(a0, k1a, zacc);
        acc.v = WMMA_F32_F16(a1, k1b, acc.v);
        store_score_tile(sc, acc, t0, (st + 1) * 16, gM, Nl, qrel_bh);
    }
    __syncthreads();

    // ---- softmax over s (two lanes per row) ----
    const int row = lane & 15;
    const int hs  = lane >> 4;
    float* rp = sc + row * T_SEQ + hs * 512;
    float mx = -3.0e38f;
#pragma unroll 8
    for (int i = 0; i < 512; ++i) mx = fmaxf(mx, rp[i]);
    mx = fmaxf(mx, __shfl_xor(mx, 16, 32));
    float sum = 0.0f;
#pragma unroll 8
    for (int i = 0; i < 512; ++i) { float e = __expf(rp[i] - mx); rp[i] = e; sum += e; }
    sum += __shfl_xor(sum, 16, 32);
    const float inv = 1.0f / sum;
#pragma unroll 8
    for (int i = 0; i < 512; ++i) rp[i] *= inv;
    __syncthreads();

    // ---- accumulate averaged weights: avg[b][t][s] += p / H ----
    {
        float* arow = avg + ((size_t)b * T_SEQ + (t0 + row)) * T_SEQ + hs * 512;
#pragma unroll 4
        for (int i = 0; i < 512; ++i) atomicAdd(&arow[i], rp[i] * (1.0f / 16.0f));
    }

    // ---- o = p · v  (v tiles async-staged through LDS, double-buffered) ----
    V8F o[4];
#pragma unroll
    for (int j = 0; j < 4; ++j) o[j].v = zacc;

    const int mA = lane & 15, g = (lane >> 4) & 1;
    stage_v_tile(vb, vtile[0], lane);
#pragma unroll 1
    for (int sk = 0; sk < 32; ++sk) {
        if (sk + 1 < 32)
            stage_v_tile(vb + (size_t)(sk + 1) * 32 * HD, vtile[(sk + 1) & 1], lane);
        wait_v_tile(sk + 1 < 32);

        // A fragment from LDS probabilities (f32 -> f16)
        V16 a;
        const float* pp = sc + mA * T_SEQ + sk * 32 + g * 8;
#pragma unroll
        for (int v = 0; v < 8; ++v) {
            int kbase = ((v & 4) ? 16 : 0) + (v & 3) * 2;
            a.e[2 * v]     = (half_t)pp[kbase];
            a.e[2 * v + 1] = (half_t)pp[kbase + 1];
        }
        const half_t* vt = vtile[sk & 1] + (size_t)(g * 8) * HD;
#pragma unroll
        for (int j = 0; j < 4; ++j) {
            V16 bf;
            const half_t* vp = vt + j * 16 + Nl;
#pragma unroll
            for (int vv = 0; vv < 8; ++vv) {
                int kbase = ((vv & 4) ? 16 : 0) + (vv & 3) * 2;
                bf.e[2 * vv]     = vp[(size_t)kbase * HD];
                bf.e[2 * vv + 1] = vp[(size_t)(kbase + 1) * HD];
            }
            o[j].v = WMMA_F32_F16(a.v, bf.v, o[j].v);
        }
    }

    // ---- store into [T,B,E] f16 layout for the output projection ----
#pragma unroll
    for (int j = 0; j < 4; ++j) {
#pragma unroll
        for (int r = 0; r < 8; ++r) {
            int t = t0 + r + gM;
            int d = j * 16 + Nl;
            attnh[((size_t)t * BSZ + b) * EMB + h * HD + d] = (half_t)o[j].e[r];
        }
    }
}

// ---------------------------------------------------------------------------
// Output projection: out[m,n] = attnh[m,:]·Wout[n,:] + bias[n]  (f32 out)
// ---------------------------------------------------------------------------
__global__ __launch_bounds__(256) void gemm_out_kernel(
    const half_t* __restrict__ A, const half_t* __restrict__ W,
    const float* __restrict__ bias, float* __restrict__ out) {
    const int lane = threadIdx.x & 31;
    const int wid  = blockIdx.x * (blockDim.x >> 5) + (threadIdx.x >> 5);
    const int TN = EMB / 64; // 16
    const int m0 = (wid / TN) * 16;
    const int n0 = (wid % TN) * 64;

    V8F acc[4];
#pragma unroll
    for (int j = 0; j < 4; ++j) acc[j].v = (v8f){0.f,0.f,0.f,0.f,0.f,0.f,0.f,0.f};

    gemm_accum_16x64(A + (size_t)m0 * EMB, W + (size_t)n0 * EMB, lane, acc);

    const int Nl = lane & 15;
    const int gM = (lane >> 4) << 3;
#pragma unroll
    for (int j = 0; j < 4; ++j) {
        int n = n0 + j * 16 + Nl;
        float bv = bias[n];
#pragma unroll
        for (int r = 0; r < 8; ++r) {
            int m = m0 + r + gM;
            out[(size_t)m * EMB + n] = acc[j].e[r] + bv;
        }
    }
}

// ---------------------------------------------------------------------------
extern "C" void kernel_launch(void* const* d_in, const int* in_sizes, int n_in,
                              void* d_out, int out_size, void* d_ws, size_t ws_size,
                              hipStream_t stream) {
    (void)in_sizes; (void)n_in; (void)out_size; (void)ws_size;
    const float* query  = (const float*)d_in[0]; // [T,B,E]
    const float* w_in   = (const float*)d_in[1]; // [3E,E]
    const float* b_in   = (const float*)d_in[2]; // [3E]
    const float* w_out  = (const float*)d_in[3]; // [E,E]
    const float* b_out  = (const float*)d_in[4]; // [E]
    const float* relk   = (const float*)d_in[5]; // [33,HD]

    float* out_attn = (float*)d_out;                                   // 8192*1024
    float* out_avg  = (float*)d_out + (size_t)M_ROWS * EMB;            // 8*1024*1024

    // workspace carve-up
    char* ws = (char*)d_ws;
    half_t* qh    = (half_t*)(ws);                       // 8192*1024
    half_t* winh  = (half_t*)(ws + 16777216);            // 3072*1024
    half_t* wouth = (half_t*)(ws + 23068672);            // 1024*1024
    half_t* qbuf  = (half_t*)(ws + 25165824);            // 128*1024*64
    half_t* kbuf  = (half_t*)(ws + 41943040);
    half_t* vbuf  = (half_t*)(ws + 58720256);
    half_t* attnh = (half_t*)(ws + 75497472);            // 8192*1024
    float*  qrel  = (float*) (ws + 92274688);            // 128*1024*33 f32

    const size_t nQ = (size_t)M_ROWS * EMB;       // 8388608
    const size_t nW = (size_t)3 * EMB * EMB;      // 3145728
    const size_t nWo = (size_t)EMB * EMB;         // 1048576
    const size_t nRel = (size_t)BSZ * H * T_SEQ * NREL;

    f32_to_f16_kernel<<<(int)((nQ + 255) / 256), 256, 0, stream>>>(query, qh, nQ);
    f32_to_f16_kernel<<<(int)((nW + 255) / 256), 256, 0, stream>>>(w_in, winh, nW);
    f32_to_f16_kernel<<<(int)((nWo + 255) / 256), 256, 0, stream>>>(w_out, wouth, nWo);

    // QKV projection: 512 x 48 wave-tiles, 8 waves/block
    gemm_qkv_kernel<<<3072, 256, 0, stream>>>(qh, winh, b_in, qbuf, kbuf, vbuf);

    qrel_kernel<<<(int)((nRel + 255) / 256), 256, 0, stream>>>(qbuf, relk, qrel);
    zero_f32_kernel<<<(int)((nQ + 255) / 256), 256, 0, stream>>>(out_avg, nQ);

    // 128 heads x 64 row-tiles, one wave each
    attn_kernel<<<128 * 64, 32, 0, stream>>>(qbuf, kbuf, vbuf, qrel, out_avg, attnh);

    // output projection: 512 x 16 wave-tiles
    gemm_out_kernel<<<1024, 256, 0, stream>>>(attnh, wouth, b_out, out_attn);
}
